// TransformerNet_85255100826186
// MI455X (gfx1250) — matrix operand order repaired and checked by reading, not compiled, not run on hardware
//
#include <hip/hip_runtime.h>
#include <cmath>

#define NNODES 10000
#define NEDGES 30000
#define NHEADS 8
#define NGRAPH 50

typedef __bf16 v16bf __attribute__((ext_vector_type(16)));
typedef float  v8f   __attribute__((ext_vector_type(8)));
typedef int    v4i   __attribute__((ext_vector_type(4)));
typedef __attribute__((address_space(1))) v4i* as1_v4i_p;   // global
typedef __attribute__((address_space(3))) v4i* as3_v4i_p;   // LDS

#if __has_builtin(__builtin_amdgcn_global_load_async_to_lds_b128) && \
    __has_builtin(__builtin_amdgcn_s_wait_asynccnt)
#define USE_ASYNC_LDS 1
#else
#define USE_ASYNC_LDS 0
#endif

__device__ inline __bf16 f2bf(float f) {
  unsigned u = __float_as_uint(f);
  unsigned r = (u + 0x7FFFu + ((u >> 16) & 1u)) >> 16;   // round-to-nearest-even
  unsigned short s = (unsigned short)r;
  return __builtin_bit_cast(__bf16, s);
}

// ordered-uint encoding so float max can use integer atomicMax (0 == identity)
__device__ inline unsigned fenc(float f) {
  unsigned b = __float_as_uint(f);
  return (b & 0x80000000u) ? ~b : (b | 0x80000000u);
}
__device__ inline float fdec(unsigned u) {
  unsigned b = (u & 0x80000000u) ? (u & 0x7FFFFFFFu) : ~u;
  return __uint_as_float(b);
}

// f32 -> bf16 bulk conversion (amortizes conversion out of the GEMM hot loop)
__global__ void f32_to_bf16(const float* __restrict__ in, __bf16* __restrict__ out, long n) {
  for (long i = (long)blockIdx.x * blockDim.x + threadIdx.x; i < n;
       i += (long)gridDim.x * blockDim.x)
    out[i] = f2bf(in[i]);
}

// ---------------------------------------------------------------------------
// Tiled bf16 WMMA GEMM:  Y[M,Nc] = Xb[M,K] * Wb[K,Nc] + bias  (bf16 in, f32 out)
// Block tile 128x128, BK=32 (one v_wmma_f32_16x16x32_bf16 K-step), double-
// buffered LDS. A tile staged with GLOBAL_LOAD_ASYNC_TO_LDS_B128 (ASYNCcnt)
// when available; B staged transposed so both fragments are contiguous per
// lane (2x ds_load_b128 each). 8 waves 4(M)x2(N); 8 WMMAs per K-step.
// ---------------------------------------------------------------------------
#define BM 128
#define BN 128
#define BK 32
#define KPAD 8      // stride 40 bf16 = 80B: 16B-aligned frag starts

__global__ __launch_bounds__(256)
void gemm_bias_wmma(const __bf16* __restrict__ X, const __bf16* __restrict__ W,
                    const float* __restrict__ bias, float* __restrict__ Y,
                    int M, int K, int Nc)
{
  __shared__ __bf16 As[2][BM][BK + KPAD];   // [row][k]
  __shared__ __bf16 Bs[2][BN][BK + KPAD];   // [col][k]  (transposed tile)
  const int t = threadIdx.x;
  const int mBase = blockIdx.y * BM;
  const int nBase = blockIdx.x * BN;
  const int wv = t >> 5, lane = t & 31;
  const int wm = wv & 3, wn = wv >> 2;           // wave grid 4 x 2
  const int lm = lane & 15, lh = lane >> 4;

  v8f acc[2][4];
#pragma unroll
  for (int i = 0; i < 2; ++i)
#pragma unroll
    for (int j = 0; j < 4; ++j)
#pragma unroll
      for (int r = 0; r < 8; ++r) acc[i][j][r] = 0.0f;

  // stage one tile pair into LDS buffer `buf`
  auto stage = [&](int buf, int k0) {
    // A tile: BM x BK bf16 = 512 16B chunks over 256 threads (2 each)
#pragma unroll
    for (int i = 0; i < 2; ++i) {
      int lin = t + i * 256;                // 0..511
      int row = lin >> 2;                   // 0..127
      int c8  = (lin & 3) << 3;             // bf16 element offset 0,8,16,24
      int gr  = mBase + row;
      if (gr < M) {
        const __bf16* ap = X + (size_t)gr * K + k0 + c8;
#if USE_ASYNC_LDS
        __builtin_amdgcn_global_load_async_to_lds_b128(
            (as1_v4i_p)(ap), (as3_v4i_p)(&As[buf][row][c8]), 0, 0);
#else
        *(uint4*)&As[buf][row][c8] = *(const uint4*)ap;
#endif
        if (k0 + 2 * BK < K) __builtin_prefetch(ap + 2 * BK, 0, 0);
      } else {
        *(uint4*)&As[buf][row][c8] = make_uint4(0u, 0u, 0u, 0u);
      }
    }
    // B tile: BK x BN bf16, stored transposed into Bs[col][k]
#pragma unroll
    for (int i = 0; i < 4; ++i) {
      int lin = t + i * 256;
      int r  = lin >> 5;                 // k within tile, 0..31
      int c4 = (lin & 31) << 2;          // col within tile
      int gc = nBase + c4;
      const __bf16* wr = W + (size_t)(k0 + r) * Nc;
      __bf16 v0 = (__bf16)0.f, v1 = v0, v2 = v0, v3 = v0;
      if (gc + 3 < Nc) {
        v0 = wr[gc + 0]; v1 = wr[gc + 1]; v2 = wr[gc + 2]; v3 = wr[gc + 3];
        if (k0 + 2 * BK < K) __builtin_prefetch(wr + 2 * (size_t)BK * Nc + gc, 0, 0);
      } else {
        if (gc + 0 < Nc) v0 = wr[gc + 0];
        if (gc + 1 < Nc) v1 = wr[gc + 1];
        if (gc + 2 < Nc) v2 = wr[gc + 2];
      }
      Bs[buf][c4 + 0][r] = v0;
      Bs[buf][c4 + 1][r] = v1;
      Bs[buf][c4 + 2][r] = v2;
      Bs[buf][c4 + 3][r] = v3;
    }
  };

  stage(0, 0);
  int cur = 0;
  for (int k0 = 0; k0 < K; k0 += BK) {
#if USE_ASYNC_LDS
    __builtin_amdgcn_s_wait_asynccnt(0);   // my async A-tile writes to LDS done
#endif
    __syncthreads();                       // everyone's staging of `cur` visible
    if (k0 + BK < K) stage(cur ^ 1, k0 + BK);   // overlap next stage with compute

    // ---- fragments in k = 16h+s order, same for A and B (both contiguous per lane)
    v16bf afrag[2], bfrag[4];
#pragma unroll
    for (int mt = 0; mt < 2; ++mt) {
      int arow = wm * 32 + mt * 16 + lm;
#pragma unroll
      for (int e = 0; e < 16; ++e) {
        int kk = e + 16 * lh;   // contiguous k = 16h+s, matches B
        afrag[mt][e] = As[cur][arow][kk];
      }
    }
#pragma unroll
    for (int nt = 0; nt < 4; ++nt) {
      int bcol = wn * 64 + nt * 16 + lm;
#pragma unroll
      for (int e = 0; e < 16; ++e)
        bfrag[nt][e] = Bs[cur][bcol][e + 16 * lh];  // contiguous -> 2x ds_load_b128
    }
#pragma unroll
    for (int mt = 0; mt < 2; ++mt)
#pragma unroll
      for (int nt = 0; nt < 4; ++nt)
        acc[mt][nt] = __builtin_amdgcn_wmma_f32_16x16x32_bf16(
            false, afrag[mt], false, bfrag[nt], (short)0, acc[mt][nt], false, false);
    cur ^= 1;
  }

  // ---- store D (+bias): M = r + 8*lh, N = lm within each 16x16 tile
#pragma unroll
  for (int mt = 0; mt < 2; ++mt)
#pragma unroll
    for (int nt = 0; nt < 4; ++nt)
#pragma unroll
      for (int r = 0; r < 8; ++r) {
        int grow = mBase + wm * 32 + mt * 16 + r + 8 * lh;
        int gcol = nBase + wn * 64 + nt * 16 + lm;
        if (grow < M && gcol < Nc)
          Y[(size_t)grow * Nc + gcol] = acc[mt][nt][r] + bias[gcol];
      }
}

// ---------------------------------------------------------------------------
__global__ void fill_f32(float* p, float v, long n) {
  for (long i = (long)blockIdx.x * blockDim.x + threadIdx.x; i < n;
       i += (long)gridDim.x * blockDim.x)
    p[i] = v;
}

// one wave32 per (edge, head): alpha = <q[dst,h,:], k[src,h,:]> / sqrt(C)
__global__ __launch_bounds__(256)
void edge_alpha(const float* __restrict__ q, const float* __restrict__ kf,
                const int* __restrict__ src, const int* __restrict__ dst,
                float* __restrict__ alpha, int C, float scale)
{
  int gw = blockIdx.x * 8 + (threadIdx.x >> 5);
  int lane = threadIdx.x & 31;
  if (gw >= NEDGES * NHEADS) return;
  int e = gw / NHEADS, hh = gw - e * NHEADS;
  const float* qp = q  + (size_t)dst[e] * (NHEADS * C) + (size_t)hh * C;
  const float* kp = kf + (size_t)src[e] * (NHEADS * C) + (size_t)hh * C;
  float sum = 0.f;
  for (int c = lane; c < C; c += 32) sum += qp[c] * kp[c];
#pragma unroll
  for (int off = 16; off; off >>= 1) sum += __shfl_xor(sum, off, 32);
  if (lane == 0) alpha[gw] = sum * scale;
}

__global__ void seg_max(const float* __restrict__ alpha, const int* __restrict__ dst,
                        unsigned* __restrict__ mx) {
  int i = blockIdx.x * blockDim.x + threadIdx.x;
  if (i >= NEDGES * NHEADS) return;
  int e = i / NHEADS, hh = i - e * NHEADS;
  atomicMax(mx + dst[e] * NHEADS + hh, fenc(alpha[i]));
}

__global__ void seg_expsum(const float* __restrict__ alpha, const int* __restrict__ dst,
                           const unsigned* __restrict__ mx, float* __restrict__ exb,
                           float* __restrict__ den) {
  int i = blockIdx.x * blockDim.x + threadIdx.x;
  if (i >= NEDGES * NHEADS) return;
  int e = i / NHEADS, hh = i - e * NHEADS;
  int d = dst[e];
  float ex = __expf(alpha[i] - fdec(mx[d * NHEADS + hh]));
  exb[i] = ex;
  atomicAdd(den + d * NHEADS + hh, ex);
}

// block per edge: agg[dst] += softmax_w * v[src]
__global__ __launch_bounds__(256)
void edge_agg(const float* __restrict__ v, const float* __restrict__ exb,
              const float* __restrict__ den, const int* __restrict__ src,
              const int* __restrict__ dst, float* __restrict__ agg, int C)
{
  __shared__ float wsh[NHEADS];
  int e = blockIdx.x;
  int d = dst[e], s = src[e];
  int hc = NHEADS * C;
  if (threadIdx.x < NHEADS)
    wsh[threadIdx.x] = exb[e * NHEADS + threadIdx.x] /
                       fmaxf(den[d * NHEADS + threadIdx.x], 1e-16f);
  __syncthreads();
  for (int c = threadIdx.x; c < hc; c += blockDim.x)
    atomicAdd(agg + (size_t)d * hc + c, wsh[c / C] * v[(size_t)s * hc + c]);
}

// out = elu( mean_heads(agg) + skip )
__global__ void finalize_elu(const float* __restrict__ agg, const float* __restrict__ skip,
                             float* __restrict__ out, int C) {
  int i = blockIdx.x * blockDim.x + threadIdx.x;
  if (i >= NNODES * C) return;
  int n = i / C, c = i - n * C;
  float s = 0.f;
#pragma unroll
  for (int h = 0; h < NHEADS; ++h) s += agg[(size_t)n * NHEADS * C + (size_t)h * C + c];
  s = s * (1.0f / NHEADS) + skip[i];
  out[i] = s > 0.f ? s : (__expf(s) - 1.f);
}

// ----- global attention pooling -----
__global__ void graph_max(const float* __restrict__ gate, const int* __restrict__ batch,
                          unsigned* __restrict__ gmx) {
  int n = blockIdx.x * blockDim.x + threadIdx.x;
  if (n >= NNODES) return;
  atomicMax(gmx + batch[n], fenc(gate[n]));
}
__global__ void graph_exp(const float* __restrict__ gate, const int* __restrict__ batch,
                          const unsigned* __restrict__ gmx, float* __restrict__ ge,
                          float* __restrict__ gden) {
  int n = blockIdx.x * blockDim.x + threadIdx.x;
  if (n >= NNODES) return;
  float v = __expf(gate[n] - fdec(gmx[batch[n]]));
  ge[n] = v;
  atomicAdd(gden + batch[n], v);
}
__global__ void graph_pool(const float* __restrict__ h, const float* __restrict__ ge,
                           const float* __restrict__ gden, const int* __restrict__ batch,
                           float* __restrict__ pool) {
  int i = blockIdx.x * blockDim.x + threadIdx.x;
  if (i >= NNODES * 64) return;
  int n = i / 64, c = i - n * 64;
  int g = batch[n];
  float wt = ge[n] / fmaxf(gden[g], 1e-16f);
  atomicAdd(pool + g * 64 + c, wt * h[i]);
}
__global__ void fc_out_kernel(const float* __restrict__ pool, const float* __restrict__ w,
                              const float* __restrict__ b, float* __restrict__ out) {
  int t = threadIdx.x;
  if (t >= NGRAPH * 10) return;
  int g = t / 10, o = t - g * 10;
  float s = b[o];
#pragma unroll
  for (int c = 0; c < 64; ++c) s += pool[g * 64 + c] * w[c * 10 + o];
  out[t] = s;
}

// ---------------------------------------------------------------------------
static void run_gemm(const __bf16* Xb, const float* Wf, __bf16* wb, const float* b,
                     float* Y, int M, int K, int Nc, hipStream_t stream) {
  f32_to_bf16<<<1024, 256, 0, stream>>>(Wf, wb, (long)K * Nc);   // convert weight once
  dim3 grid((Nc + BN - 1) / BN, (M + BM - 1) / BM);
  gemm_bias_wmma<<<grid, 256, 0, stream>>>(Xb, wb, b, Y, M, K, Nc);
}

static void run_tconv(const float* x, int fin, int cout,
                      const float* qw, const float* qb, const float* kw, const float* kb,
                      const float* vw, const float* vb, const float* sw, const float* sb,
                      const int* src, const int* dst,
                      float* q, float* kbuf, float* v, float* alpha, float* exb,
                      unsigned* mx, float* den, __bf16* xb, __bf16* wb,
                      float* out, hipStream_t stream)
{
  const int C = cout, hc = NHEADS * cout;
  const int EH = NEDGES * NHEADS;
  const float scale = 1.0f / sqrtf((float)C);
  float* skip = kbuf;        // k is dead after edge_alpha
  float* agg  = q;           // q is dead after edge_alpha

  // convert layer input to bf16 once; shared by the q/k/v/skip projections
  f32_to_bf16<<<1024, 256, 0, stream>>>(x, xb, (long)NNODES * fin);

  run_gemm(xb, qw, wb, qb, q,    NNODES, fin, hc, stream);
  run_gemm(xb, kw, wb, kb, kbuf, NNODES, fin, hc, stream);
  edge_alpha<<<(EH + 7) / 8, 256, 0, stream>>>(q, kbuf, src, dst, alpha, C, scale);

  fill_f32<<<256, 256, 0, stream>>>((float*)mx, 0.f, (long)NNODES * NHEADS);
  seg_max<<<(EH + 255) / 256, 256, 0, stream>>>(alpha, dst, mx);
  fill_f32<<<256, 256, 0, stream>>>(den, 0.f, (long)NNODES * NHEADS);
  seg_expsum<<<(EH + 255) / 256, 256, 0, stream>>>(alpha, dst, mx, exb, den);

  run_gemm(xb, vw, wb, vb, v,    NNODES, fin, hc, stream);
  run_gemm(xb, sw, wb, sb, skip, NNODES, fin, cout, stream);

  fill_f32<<<2048, 256, 0, stream>>>(agg, 0.f, (long)NNODES * hc);
  edge_agg<<<NEDGES, 256, 0, stream>>>(v, exb, den, src, dst, agg, C);
  finalize_elu<<<(NNODES * cout + 255) / 256, 256, 0, stream>>>(agg, skip, out, cout);
}

extern "C" void kernel_launch(void* const* d_in, const int* in_sizes, int n_in,
                              void* d_out, int out_size, void* d_ws, size_t ws_size,
                              hipStream_t stream)
{
  (void)in_sizes; (void)n_in; (void)out_size; (void)ws_size;
  const float* x     = (const float*)d_in[0];
  const int*   ei    = (const int*)d_in[1];
  const int*   batch = (const int*)d_in[2];
  const int* src = ei;
  const int* dst = ei + NEDGES;

  float* ws = (float*)d_ws;
  const size_t NQ = (size_t)NNODES * 4096;        // max hc (layer 1)
  float* q     = ws;
  float* kbuf  = ws + NQ;
  float* v     = ws + 2 * NQ;
  float* hA    = ws + 3 * NQ;                      // up to N x 512
  float* hB    = hA + (size_t)NNODES * 512;
  float* alpha = hB + (size_t)NNODES * 512;
  float* exb   = alpha + (size_t)NEDGES * NHEADS;
  unsigned* mx = (unsigned*)(exb + (size_t)NEDGES * NHEADS);
  float* den   = (float*)mx + (size_t)NNODES * NHEADS;
  float* gate  = den + (size_t)NNODES * NHEADS;
  float* ge    = gate + NNODES;
  unsigned* gmx = (unsigned*)(ge + NNODES);
  float* gden  = (float*)gmx + 64;
  float* pool  = gden + 64;
  __bf16* xb   = (__bf16*)(pool + (size_t)NGRAPH * 64);   // N x 512 bf16 max
  __bf16* wb   = xb + (size_t)NNODES * 512;               // 512 x 4096 bf16 max

  // layer parameter slots: 3 + 8*(layer-1)
  const float** p = (const float**)d_in;
  run_tconv(x, 128, 512, p[3], p[4], p[5], p[6], p[7], p[8], p[9], p[10],
            src, dst, q, kbuf, v, alpha, exb, mx, den, xb, wb, hA, stream);
  run_tconv(hA, 512, 256, p[11], p[12], p[13], p[14], p[15], p[16], p[17], p[18],
            src, dst, q, kbuf, v, alpha, exb, mx, den, xb, wb, hB, stream);
  run_tconv(hB, 256, 64, p[19], p[20], p[21], p[22], p[23], p[24], p[25], p[26],
            src, dst, q, kbuf, v, alpha, exb, mx, den, xb, wb, hA, stream);

  // global attention pooling over graphs
  f32_to_bf16<<<1024, 256, 0, stream>>>(hA, xb, (long)NNODES * 64);
  run_gemm(xb, p[27], wb, p[28], gate, NNODES, 64, 1, stream);  // gate = h @ gate_w + b
  fill_f32<<<1, 64, 0, stream>>>((float*)gmx, 0.f, 64);
  fill_f32<<<1, 64, 0, stream>>>(gden, 0.f, 64);
  fill_f32<<<16, 256, 0, stream>>>(pool, 0.f, (long)NGRAPH * 64);
  graph_max<<<(NNODES + 255) / 256, 256, 0, stream>>>(gate, batch, gmx);
  graph_exp<<<(NNODES + 255) / 256, 256, 0, stream>>>(gate, batch, gmx, ge, gden);
  graph_pool<<<(NNODES * 64 + 255) / 256, 256, 0, stream>>>(hA, ge, gden, batch, pool);
  fc_out_kernel<<<1, 512, 0, stream>>>(pool, p[29], p[30], (float*)d_out);
}
